// SelfAttentionV1_10393820857170
// MI455X (gfx1250) — matrix operand order repaired and checked by reading, not compiled, not run on hardware
//
#include <hip/hip_runtime.h>
#include <hip/hip_bf16.h>
#include <math.h>

typedef __attribute__((ext_vector_type(16))) _Float16 v16h;
typedef __attribute__((ext_vector_type(8)))  _Float16 v8h;
typedef __attribute__((ext_vector_type(8)))  float    v8f;

#define HIDDEN 768
#define SEQ    2048
#define BATCH  4
#define SCALE  0.0360843918243516f   /* 1/sqrt(768) */

__device__ __forceinline__ v8f wmma_f16(v16h a, v16h b, v8f c) {
  // D = A(16x32 f16) * B(32x16 f16) + C(16x16 f32)
  return __builtin_amdgcn_wmma_f32_16x16x32_f16(false, a, false, b,
                                                (short)0, c, false, false);
}

// ---------------------------------------------------------------------------
// Kernel 1: Q[m,n] = sum_k X[m,k] * Wq[n,k] + bq[n].
// Emits TWO f16 copies into ws:
//   Qh[m][n]      (row-major over hidden)  -> used for Q/K fragments (scores)
//   Qt[b][n][s]   (transposed, key-major)  -> used for V fragments (P@V),
//                 making every V B-fragment a single contiguous 32B load.
// One 16x16 tile per wave. CDNA5 wave32 WMMA layouts:
//   A: lane%16 = M ; halves h -> K = (h<8 ? h : h+8) + 8*g
//   B: lane%16 = N ; halves h -> K = (h<8 ? h : h+8) + 8*g
//   C/D f32: lane%16 = N ; reg r -> M = r + 8*g
// The D layout gives each lane 8 consecutive M (=seq) values, so the Qt
// (transposed) write is one aligned v8h (16B) store per lane.
// ---------------------------------------------------------------------------
__global__ __launch_bounds__(256)
void proj_q_kernel(const float* __restrict__ X, const float* __restrict__ W,
                   const float* __restrict__ bias,
                   _Float16* __restrict__ Qh, _Float16* __restrict__ Qt)
{
  const int lane = threadIdx.x & 31;
  const int wave = threadIdx.x >> 5;
  const int g    = lane >> 4;
  const int ln   = lane & 15;

  const int tile = blockIdx.x * 8 + wave;        // 8 waves -> 8 tiles per block
  const int m0   = (tile / (HIDDEN / 16)) * 16;  // row tile (batch*seq)
  const int n0   = (tile % (HIDDEN / 16)) * 16;  // col tile (hidden)

  const float* xrow = X + (size_t)(m0 + ln) * HIDDEN;
  const float* wrow = W + (size_t)(n0 + ln) * HIDDEN;

  v8f acc = {};
  for (int k0 = 0; k0 < HIDDEN; k0 += 32) {
    v8f a0 = *(const v8f*)(xrow + k0 + 8 * g);
    v8f a1 = *(const v8f*)(xrow + k0 + 8 * g + 16);
    v8f b0 = *(const v8f*)(wrow + k0 + 8 * g);
    v8f b1 = *(const v8f*)(wrow + k0 + 8 * g + 16);
    v16h a, b;
#pragma unroll
    for (int h = 0; h < 8; ++h) {
      a[h]     = (_Float16)a0[h];
      a[h + 8] = (_Float16)a1[h];
      b[h]     = (_Float16)b0[h];
      b[h + 8] = (_Float16)b1[h];
    }
    acc = wmma_f16(a, b, acc);
  }

  const float bn = bias[n0 + ln];
  _Float16 hv[8];
#pragma unroll
  for (int r = 0; r < 8; ++r) hv[r] = (_Float16)(acc[r] + bn);

  // Row-major copy: lane has fixed col n0+ln, 8 strided rows.
#pragma unroll
  for (int r = 0; r < 8; ++r) {
    const int m = m0 + r + 8 * g;
    Qh[(size_t)m * HIDDEN + n0 + ln] = hv[r];
  }

  // Transposed copy: Qt[b][col][seq]; 8 consecutive seq values per lane.
  {
    const int bB = m0 >> 11;          // m0 / 2048 (tiles never straddle batch)
    const int s0 = (m0 & 2047) + 8 * g;
    v8h pack;
#pragma unroll
    for (int r = 0; r < 8; ++r) pack[r] = hv[r];
    *(v8h*)(Qt + ((size_t)bB * HIDDEN + n0 + ln) * SEQ + s0) = pack;
  }
}

// ---------------------------------------------------------------------------
// Kernel 2: single-pass flash attention with Q == K == V.
// Block = (batch, 16-query tile), 8 wave32 waves.
// Per 128-key block:
//   Phase A: wave w computes the 16x16 score tile for keys [kb+16w, +16)
//            over the full 768 reduction (24 WMMAs) from Qh (contiguous).
//   Phase B: cross-wave online softmax (running row max/sum); P -> LDS f16.
//   Phase C: every wave rescales its private O slice (96 cols, 6 f32 tiles)
//            and accumulates P @ V; V fragments are single v16h loads from Qt.
// Epilogue: O /= row_sum, store fp32.
// ---------------------------------------------------------------------------
__global__ __launch_bounds__(256)
void attn_kernel(const _Float16* __restrict__ Q, const _Float16* __restrict__ Qt,
                 float* __restrict__ out)
{
  __shared__ float    sS[16][128];
  __shared__ _Float16 sP[16][128];
  __shared__ float    sM[16], sL[16], sA[16];

  const int lane = threadIdx.x & 31;
  const int wave = threadIdx.x >> 5;
  const int g    = lane >> 4;
  const int ln   = lane & 15;

  const int b  = blockIdx.x >> 7;            // 128 query tiles per batch
  const int q0 = (blockIdx.x & 127) * 16;
  const _Float16* Qb  = Q  + (size_t)b * SEQ * HIDDEN;
  const _Float16* Qtb = Qt + (size_t)b * HIDDEN * SEQ;

  if (threadIdx.x < 16) { sM[threadIdx.x] = -3.0e38f; sL[threadIdx.x] = 0.0f; }

  v8f O[6];
#pragma unroll
  for (int t = 0; t < 6; ++t) O[t] = (v8f){};

  const int colBase = wave * 96;
  const _Float16* qrow = Qb + (size_t)(q0 + ln) * HIDDEN;

  for (int kb = 0; kb < SEQ; kb += 128) {
    // -------- Phase A: scores for this wave's 16 keys (registers only) -----
    const _Float16* krow = Qb + (size_t)(kb + wave * 16 + ln) * HIDDEN;
    if (kb + 128 < SEQ)                         // warm L2/L0 for next block
      __builtin_prefetch(krow + (size_t)128 * HIDDEN, 0, 1);

    v8f s = {};
    for (int kc = 0; kc < HIDDEN; kc += 32) {
      v8h a0 = *(const v8h*)(qrow + kc + 16 * g);
      v8h a1 = *(const v8h*)(qrow + kc + 16 * g + 8);
      v16h a = __builtin_shufflevector(a0, a1,
               0, 1, 2, 3, 4, 5, 6, 7, 8, 9, 10, 11, 12, 13, 14, 15);
      v16h bf = *(const v16h*)(krow + kc + 16 * g);
      s = wmma_f16(a, bf, s);
    }

    __syncthreads();  // previous iteration's sP/sA consumers are done
#pragma unroll
    for (int r = 0; r < 8; ++r)
      sS[r + 8 * g][wave * 16 + ln] = s[r] * SCALE;
    __syncthreads();

    // -------- Phase B: online softmax; wave handles rows {2w, 2w+1} --------
    {
      const int row = 2 * wave + g;
      float v[8];
      float mx = -3.0e38f;
#pragma unroll
      for (int i = 0; i < 8; ++i) {
        v[i] = sS[row][ln + 16 * i];
        mx = fmaxf(mx, v[i]);
      }
#pragma unroll
      for (int off = 8; off >= 1; off >>= 1)
        mx = fmaxf(mx, __shfl_xor(mx, off, 16));

      const float oldM = sM[row];
      const float newM = fmaxf(oldM, mx);
      float sum = 0.0f;
#pragma unroll
      for (int i = 0; i < 8; ++i) {
        const float p = __expf(v[i] - newM);
        sum += p;
        sP[row][ln + 16 * i] = (_Float16)p;
      }
#pragma unroll
      for (int off = 8; off >= 1; off >>= 1)
        sum += __shfl_xor(sum, off, 16);

      if (ln == 0) {
        const float alpha = __expf(oldM - newM);
        sA[row] = alpha;
        sM[row] = newM;
        sL[row] = sL[row] * alpha + sum;
      }
    }
    __syncthreads();

    // -------- Phase C: rescale O, accumulate P @ V -------------------------
    float alpha_r[8];
#pragma unroll
    for (int r = 0; r < 8; ++r) alpha_r[r] = sA[r + 8 * g];
#pragma unroll
    for (int t = 0; t < 6; ++t)
#pragma unroll
      for (int r = 0; r < 8; ++r) O[t][r] *= alpha_r[r];

    for (int ks = 0; ks < 128; ks += 32) {
      const _Float16* prow = &sP[ln][ks + 16 * g];
      v8h p0 = *(const v8h*)(prow);
      v8h p1 = *(const v8h*)(prow + 8);
      v16h a = __builtin_shufflevector(p0, p1,
               0, 1, 2, 3, 4, 5, 6, 7, 8, 9, 10, 11, 12, 13, 14, 15);
#pragma unroll
      for (int t = 0; t < 6; ++t) {
        // V B-fragment from the transposed copy: lane = col, halves = 16
        // contiguous keys -> one aligned 32B vector load.
        const _Float16* vp =
            Qtb + (size_t)(colBase + 16 * t + ln) * SEQ + kb + ks + 16 * g;
        v16h bf = *(const v16h*)(vp);
        O[t] = wmma_f16(a, bf, O[t]);
      }
    }
  }

  // -------- Epilogue: normalize by row sums and store fp32 -----------------
  float linv[8];
#pragma unroll
  for (int r = 0; r < 8; ++r) linv[r] = 1.0f / sL[r + 8 * g];

  float* outB = out + ((size_t)b * SEQ + q0) * HIDDEN;
#pragma unroll
  for (int t = 0; t < 6; ++t)
#pragma unroll
    for (int r = 0; r < 8; ++r)
      outB[(size_t)(r + 8 * g) * HIDDEN + colBase + 16 * t + ln] =
          O[t][r] * linv[r];
}

extern "C" void kernel_launch(void* const* d_in, const int* in_sizes, int n_in,
                              void* d_out, int out_size, void* d_ws, size_t ws_size,
                              hipStream_t stream) {
  (void)in_sizes; (void)n_in; (void)out_size; (void)ws_size;
  const float* X  = (const float*)d_in[0];   // [4, 2048, 768] fp32
  const float* Wq = (const float*)d_in[1];   // [768, 768]     fp32
  const float* bq = (const float*)d_in[2];   // [768]          fp32
  float*    out   = (float*)d_out;           // [4, 2048, 768] fp32

  // ws layout: Qh (row-major f16, 12.6 MB) | Qt (transposed f16, 12.6 MB)
  _Float16* Qh = (_Float16*)d_ws;
  _Float16* Qt = Qh + (size_t)BATCH * SEQ * HIDDEN;

  // Projection: 512 x 48 = 24576 tiles, 8 tiles (waves) per 256-thread block.
  const int nTilesTotal = (BATCH * SEQ / 16) * (HIDDEN / 16);
  proj_q_kernel<<<nTilesTotal / 8, 256, 0, stream>>>(X, Wq, bq, Qh, Qt);

  // Attention: one block per (batch, 16-query tile).
  attn_kernel<<<BATCH * (SEQ / 16), 256, 0, stream>>>(Qh, Qt, out);
}
